// PointNetSetAbstractionWavelet_16045997818017
// MI455X (gfx1250) — matrix operand order repaired and checked by reading, not compiled, not run on hardware
//
#include <hip/hip_runtime.h>
#include <hip/hip_bf16.h>
#include <math.h>

// ---------------------------------------------------------------------------
// Problem constants (from reference)
// ---------------------------------------------------------------------------
#define B_     8
#define N_     4096
#define CIN_   64
#define S_     1024
#define K_     32
#define C_     128
#define J_     5
#define DEPTH_ 2
#define H_     4
#define P_     (B_*S_)        // 8192 patches
#define ROWS_  (P_*K_)        // 262144 rows through the MLP
#define BETA_  0.05f
#define EPS_   1e-5f
#define CINP_  96             // 67 padded up to 3 k-tiles of 32
#define NBLK_  1024           // stats partial blocks

typedef __bf16 bf16;
typedef __attribute__((ext_vector_type(16))) __bf16 v16bf;
typedef __attribute__((ext_vector_type(8)))  float  v8f;

// ---------------------------------------------------------------------------
// CDNA5 async global->LDS staging (ASYNCcnt path, ISA ch.10 / §15.18)
// ---------------------------------------------------------------------------
__device__ __forceinline__ void async_ld_b128(uint32_t ldsByteAddr,
                                              uint64_t gAddr) {
  asm volatile("global_load_async_to_lds_b128 %0, %1, off"
               :
               : "v"(ldsByteAddr), "v"(gAddr)
               : "memory");
}
__device__ __forceinline__ void wait_async0() {
  asm volatile("s_wait_asynccnt 0x0" ::: "memory");
}

// ---------------------------------------------------------------------------
// WMMA helpers (CDNA5 wave32, D = A(16x32 bf16) x B(32x16 bf16) + C(16x16 f32))
// A layout per lane L: m = L%16, g = L/16; element e: K = e + 8g + 8*(e/8)
//   -> two contiguous 8-element (16B) runs at K = 8g and K = 16+8g
// B layout per lane L: n = L%16, g = L/16; element e: K = e + 8g + 8*(e/8)
//   -> two contiguous 8-element (16B) runs at K = 8g and K = 16+8g; B stored [N][K]
// C layout per lane L: element r -> row m = r + 8g, col n = L%16
// ---------------------------------------------------------------------------
__device__ __forceinline__ v8f wmma_bf16(v16bf a, v16bf b, v8f c) {
  return __builtin_amdgcn_wmma_f32_16x16x32_bf16(false, a, false, b, (short)0, c,
                                                 false, false);
}

__device__ __forceinline__ v16bf ldA16(const bf16* base, int stride, int m0,
                                       int k0, int lane) {
  int m = m0 + (lane & 15);
  int g = lane >> 4;
  const bf16* p0 = base + m * stride + k0 + 8 * g;
  union { v16bf v; uint4 q[2]; } u;
  u.q[0] = *(const uint4*)(p0);
  u.q[1] = *(const uint4*)(p0 + 16);
  return u.v;
}

__device__ __forceinline__ v16bf ldB16(const bf16* base, int stride, int n0,
                                       int k0, int lane) {
  int n = n0 + (lane & 15);
  int g = lane >> 4;
  const bf16* p = base + n * stride + k0 + 8 * g;
  union { v16bf v; uint4 q[2]; } u;
  u.q[0] = *(const uint4*)(p);
  u.q[1] = *(const uint4*)(p + 16);
  return u.v;
}

// A-tile load with per-input-channel BN affine + ReLU applied on the fly.
__device__ __forceinline__ v16bf ldA16_affine(const bf16* X, int stride, int m0,
                                              int k0, int lane,
                                              const float* scale,
                                              const float* bias) {
  int m = m0 + (lane & 15);
  int g = lane >> 4;
  const bf16* row = X + m * stride;
  v16bf v;
#pragma unroll
  for (int e = 0; e < 16; ++e) {
    int kk = k0 + 8 * g + ((e < 8) ? e : (e + 8));
    float f = (float)row[kk];
    f = fmaxf(f * scale[kk] + bias[kk], 0.f);
    v[e] = (bf16)f;
  }
  return v;
}

// B-tile gathered from a row-major [K][N] store (transposed access).
__device__ __forceinline__ v16bf ldB16_trans(const bf16* base, int stride,
                                             int n0, int k0, int lane) {
  int c = n0 + (lane & 15);
  int g = lane >> 4;
  v16bf v;
#pragma unroll
  for (int e = 0; e < 16; ++e) v[e] = base[(k0 + 8 * g + ((e < 8) ? e : (e + 8))) * stride + c];
  return v;
}

// ---------------------------------------------------------------------------
// Weight prep: fp32 -> bf16, padded / transposed into [out][in] layouts
// ---------------------------------------------------------------------------
__global__ void k_prep_pad(const float* __restrict__ src, bf16* __restrict__ dst,
                           int rows, int cin, int cinp) {
  int i = blockIdx.x * 256 + threadIdx.x;
  if (i >= rows * cinp) return;
  int r = i / cinp, c = i % cinp;
  dst[i] = (c < cin) ? (bf16)src[r * cin + c] : (bf16)0.f;
}

__global__ void k_prep_T(const float* __restrict__ src, bf16* __restrict__ dst,
                         int mats, int Kd, int Nd) {
  int i = blockIdx.x * 256 + threadIdx.x;
  if (i >= mats * Kd * Nd) return;
  int m = i / (Kd * Nd);
  int rem = i % (Kd * Nd);
  int o = rem / Kd, c = rem % Kd;
  dst[i] = (bf16)src[(m * Kd + c) * Nd + o];   // dst[m][o][c] = src[m][c][o]
}

// ---------------------------------------------------------------------------
// Farthest point sampling: one block per batch, 1024 threads, dist in regs.
// ---------------------------------------------------------------------------
__global__ void __launch_bounds__(1024) k_fps(const float* __restrict__ xyz,
                                              int* __restrict__ fpsIdx,
                                              float* __restrict__ newXyz) {
  __shared__ float sv[1024];
  __shared__ int   si[1024];
  __shared__ float sc[3];
  __shared__ int   sFar;
  int b = blockIdx.x, t = threadIdx.x;
  const float* xb = xyz + (size_t)b * N_ * 3;
  float dist[4];
#pragma unroll
  for (int i = 0; i < 4; ++i) dist[i] = 1e10f;
  if (t == 0) sFar = 0;
  __syncthreads();
  for (int s = 0; s < S_; ++s) {
    int far = sFar;
    if (t == 0) {
      fpsIdx[b * S_ + s] = far;
      sc[0] = xb[far * 3 + 0];
      sc[1] = xb[far * 3 + 1];
      sc[2] = xb[far * 3 + 2];
    }
    __syncthreads();
    if (t < 3) newXyz[((size_t)b * S_ + s) * 3 + t] = sc[t];
    float cx = sc[0], cy = sc[1], cz = sc[2];
    float best = -1.f;
    int bi = 0;
#pragma unroll
    for (int i = 0; i < 4; ++i) {
      int n = t + i * 1024;
      float dx = xb[n * 3 + 0] - cx, dy = xb[n * 3 + 1] - cy,
            dz = xb[n * 3 + 2] - cz;
      float d = dx * dx + dy * dy + dz * dz;
      dist[i] = fminf(dist[i], d);
      if (dist[i] > best) { best = dist[i]; bi = n; }
    }
    sv[t] = best; si[t] = bi;
    __syncthreads();
    for (int off = 512; off > 0; off >>= 1) {
      if (t < off) {
        float v2 = sv[t + off]; int i2 = si[t + off];
        if (v2 > sv[t] || (v2 == sv[t] && i2 < si[t])) { sv[t] = v2; si[t] = i2; }
      }
      __syncthreads();
    }
    if (t == 0) sFar = si[0];
    __syncthreads();
  }
}

// ---------------------------------------------------------------------------
// kNN (32 nearest of 4096) per center: iterative argmin with masking in LDS.
// ---------------------------------------------------------------------------
__global__ void __launch_bounds__(256) k_knn(const float* __restrict__ xyz,
                                             const float* __restrict__ newXyz,
                                             int* __restrict__ knn) {
  __shared__ float sd[N_];
  __shared__ float rv[256];
  __shared__ int   ri[256];
  int ps = blockIdx.x;
  int b = ps / S_;
  int t = threadIdx.x;
  const float* xb = xyz + (size_t)b * N_ * 3;
  float cx = newXyz[ps * 3 + 0], cy = newXyz[ps * 3 + 1],
        cz = newXyz[ps * 3 + 2];
  for (int n = t; n < N_; n += 256) {
    float dx = xb[n * 3 + 0] - cx, dy = xb[n * 3 + 1] - cy,
          dz = xb[n * 3 + 2] - cz;
    sd[n] = dx * dx + dy * dy + dz * dz;
  }
  __syncthreads();
  for (int r = 0; r < K_; ++r) {
    float best = 1e30f; int bi = 0;
    for (int n = t; n < N_; n += 256) {
      float v = sd[n];
      if (v < best) { best = v; bi = n; }
    }
    rv[t] = best; ri[t] = bi;
    __syncthreads();
    for (int off = 128; off > 0; off >>= 1) {
      if (t < off) {
        float v2 = rv[t + off]; int i2 = ri[t + off];
        if (v2 < rv[t] || (v2 == rv[t] && i2 < ri[t])) { rv[t] = v2; ri[t] = i2; }
      }
      __syncthreads();
    }
    if (t == 0) { knn[(size_t)ps * K_ + r] = ri[0]; sd[ri[0]] = 1e30f; }
    __syncthreads();
  }
}

// ---------------------------------------------------------------------------
// Gather grouped features: h0 = [gnorm(3) | points(64) | pad(29)] as bf16.
// ---------------------------------------------------------------------------
__global__ void k_gather(const float* __restrict__ xyz,
                         const float* __restrict__ points,
                         const float* __restrict__ newXyz,
                         const int* __restrict__ knn,
                         bf16* __restrict__ h0, float* __restrict__ gnorm) {
  int row = blockIdx.x * 256 + threadIdx.x;
  if (row >= ROWS_) return;
  int p = row >> 5;
  int b = p / S_;
  int i = knn[row];
  const float* pt = xyz + ((size_t)b * N_ + i) * 3;
  const float* c  = newXyz + (size_t)p * 3;
  float gx = pt[0] - c[0], gy = pt[1] - c[1], gz = pt[2] - c[2];
  float* gn = gnorm + (size_t)row * 3;
  gn[0] = gx; gn[1] = gy; gn[2] = gz;
  bf16* h = h0 + (size_t)row * CINP_;
  h[0] = (bf16)gx; h[1] = (bf16)gy; h[2] = (bf16)gz;
  const float* fp = points + ((size_t)b * N_ + i) * CIN_;
#pragma unroll
  for (int c2 = 0; c2 < CIN_; ++c2) h[3 + c2] = (bf16)fp[c2];
#pragma unroll
  for (int c2 = 3 + CIN_; c2 < CINP_; ++c2) h[c2] = (bf16)0.f;
}

// ---------------------------------------------------------------------------
// MLP GEMM: Y[32-row block][cout] = bnrelu(X) @ W^T, WMMA bf16, f32 accum.
// The 32-row X block is async-staged into LDS (ASYNCcnt), A-tiles read from
// LDS; W is [cout][ldx] bf16 and stays L2-hot.
// ---------------------------------------------------------------------------
__global__ void __launch_bounds__(128) k_gemm(const bf16* __restrict__ X, int ldx,
                                              const bf16* __restrict__ W, int cout,
                                              int ktiles, bf16* __restrict__ Y,
                                              int useAffine,
                                              const float* __restrict__ scale,
                                              const float* __restrict__ bias) {
  __shared__ __align__(16) bf16 sXblk[32 * CINP_];
  int wave = threadIdx.x >> 5, lane = threadIdx.x & 31;
  int row0 = blockIdx.x * 32;
  int ntiles = cout >> 4;
  const bf16* Xb = X + (size_t)row0 * ldx;

  // async stage: 32*ldx bf16 in 16B chunks (chunk count is a multiple of 128)
  int chunks = (32 * ldx) >> 3;
  for (int ch = threadIdx.x; ch < chunks; ch += 128) {
    uint32_t lo = (uint32_t)(uintptr_t)(&sXblk[ch * 8]);
    uint64_t ga = (uint64_t)(uintptr_t)(Xb + ch * 8);
    async_ld_b128(lo, ga);
  }
  wait_async0();
  __syncthreads();

  for (int t = wave; t < 2 * ntiles; t += 4) {
    int mt = t & 1, nt = t >> 1;
    v8f acc = {};
    for (int kt = 0; kt < ktiles; ++kt) {
      v16bf a = useAffine
                    ? ldA16_affine(sXblk, ldx, mt * 16, kt * 32, lane, scale, bias)
                    : ldA16(sXblk, ldx, mt * 16, kt * 32, lane);
      v16bf bm = ldB16(W, ldx, nt * 16, kt * 32, lane);
      acc = wmma_bf16(a, bm, acc);
    }
    int col = nt * 16 + (lane & 15);
    int g = lane >> 4;
#pragma unroll
    for (int r = 0; r < 8; ++r) {
      int m = row0 + mt * 16 + r + 8 * g;
      Y[(size_t)m * cout + col] = (bf16)acc[r];
    }
  }
}

// ---------------------------------------------------------------------------
// Deterministic per-channel stats: stage-1 partials per block, stage-2 in order.
// ---------------------------------------------------------------------------
__global__ void __launch_bounds__(256) k_stats(const bf16* __restrict__ Y,
                                               int rows, int cols,
                                               float* __restrict__ psum,
                                               float* __restrict__ psq) {
  __shared__ float s1[256], s2[256];
  int t = threadIdx.x;
  int c = t % cols;
  int gsz = 256 / cols;
  int gidx = t / cols;
  float a = 0.f, b = 0.f;
  for (int r = blockIdx.x * gsz + gidx; r < rows; r += gridDim.x * gsz) {
    float v = (float)Y[(size_t)r * cols + c];
    a += v; b += v * v;
  }
  s1[t] = a; s2[t] = b;
  __syncthreads();
  for (int off = 128; off >= cols; off >>= 1) {
    if (t < off) { s1[t] += s1[t + off]; s2[t] += s2[t + off]; }
    __syncthreads();
  }
  if (t < cols) {
    psum[(size_t)blockIdx.x * cols + t] = s1[t];
    psq[(size_t)blockIdx.x * cols + t]  = s2[t];
  }
}

__global__ void k_bnparam(const float* __restrict__ psum,
                          const float* __restrict__ psq, int nblk,
                          const float* __restrict__ g,
                          const float* __restrict__ b, int cols, float invM,
                          float* __restrict__ scale, float* __restrict__ bias) {
  int c = threadIdx.x;
  if (c >= cols) return;
  float s = 0.f, q = 0.f;
  for (int i = 0; i < nblk; ++i) {
    s += psum[(size_t)i * cols + c];
    q += psq[(size_t)i * cols + c];
  }
  float mean = s * invM;
  float var = q * invM - mean * mean;
  float sc = g[c] * rsqrtf(var + EPS_);
  scale[c] = sc;
  bias[c] = b[c] - mean * sc;
}

// ---------------------------------------------------------------------------
// Per-patch wavelet operators: psi_j = I + M + M^2/2 + M^3/6 + M^4/24 (bf16 out)
// One block per patch, thread (l,k) owns element (k,l); each wave is one row.
// ---------------------------------------------------------------------------
__global__ void __launch_bounds__(1024) k_psi(const float* __restrict__ gnorm,
                                              bf16* __restrict__ psi) {
  __shared__ float px[K_ * 3];
  __shared__ float Am[K_][K_ + 1];
  __shared__ float Lm[K_][K_ + 1];
  __shared__ float Mm[K_][K_ + 1];
  __shared__ float accA[K_][K_ + 1];
  __shared__ float accB[K_][K_ + 1];
  __shared__ float red[1024];
  __shared__ float dinv[K_];
  int p = blockIdx.x;
  int l = threadIdx.x, k = threadIdx.y;
  int t = k * 32 + l;
  if (t < 96) px[t] = gnorm[(size_t)p * 96 + t];
  __syncthreads();
  float dx = px[k * 3 + 0] - px[l * 3 + 0];
  float dy = px[k * 3 + 1] - px[l * 3 + 1];
  float dz = px[k * 3 + 2] - px[l * 3 + 2];
  float dd = dx * dx + dy * dy + dz * dz;
  red[t] = sqrtf(dd + 1e-12f);
  __syncthreads();
  for (int off = 512; off > 0; off >>= 1) {
    if (t < off) red[t] += red[t + off];
    __syncthreads();
  }
  float sigma = red[0] * (1.f / 1024.f);
  __syncthreads();
  float a = __expf(-dd / (2.f * sigma * sigma + 1e-12f));
  Am[k][l] = a;
  red[t] = a;
  __syncthreads();
  for (int off = 16; off > 0; off >>= 1) {
    if (l < off) red[t] += red[t + off];
    __syncthreads();
  }
  if (l == 0) dinv[k] = rsqrtf(red[k * 32] + 1e-12f);
  __syncthreads();
  Lm[k][l] = ((k == l) ? 1.f : 0.f) - dinv[k] * Am[k][l] * dinv[l];
  __syncthreads();
  for (int j = 0; j < J_; ++j) {
    float sc = -BETA_ * (float)(1 << j);
    float m0 = sc * Lm[k][l];
    Mm[k][l] = m0;
    accA[k][l] = m0;
    float T = ((k == l) ? 1.f : 0.f) + m0;
    __syncthreads();
    for (int m = 2; m <= 4; ++m) {
      float s = 0.f;
#pragma unroll
      for (int i = 0; i < K_; ++i) s += accA[k][i] * Mm[i][l];
      s *= (1.f / (float)m);
      accB[k][l] = s;
      T += s;
      __syncthreads();
      accA[k][l] = accB[k][l];
      __syncthreads();
    }
    psi[((size_t)p * J_ + j) * 1024 + k * 32 + l] = (bf16)T;
    __syncthreads();
  }
}

// ---------------------------------------------------------------------------
// Per-patch transformer (DEPTH=2): all matmuls via WMMA bf16, state in LDS.
// Block = 128 threads = 4 waves; 32x128 hcur resident; <64KB LDS.
// sScr (10240B) is time-shared: psi staging (mix phase, async-loaded) and
// attention scores (attention phase).
// ---------------------------------------------------------------------------
#define HS  132   // f32 hcur row stride (bank-decorrelated)
#define HBS 136   // bf16 row stride (16B-aligned: 272 bytes)
#define VTS 40    // 32+8 bf16 stride (80 bytes, 16B-aligned)

__device__ __forceinline__ void layer_norm(float* sH, bf16* sHb, float* sMean,
                                           float* sRstd,
                                           const float* __restrict__ g,
                                           const float* __restrict__ b,
                                           int tid) {
  if (tid < K_) {
    const float* row = sH + tid * HS;
    float s = 0.f;
#pragma unroll
    for (int c = 0; c < C_; ++c) s += row[c];
    float mu = s * (1.f / C_);
    float v = 0.f;
#pragma unroll
    for (int c = 0; c < C_; ++c) { float d = row[c] - mu; v += d * d; }
    sMean[tid] = mu;
    sRstd[tid] = rsqrtf(v * (1.f / C_) + EPS_);
  }
  __syncthreads();
  for (int idx = tid; idx < K_ * C_; idx += 128) {
    int r = idx >> 7, c = idx & 127;
    float v = (sH[r * HS + c] - sMean[r]) * sRstd[r] * g[c] + b[c];
    sH[r * HS + c] = v;
    sHb[r * HBS + c] = (bf16)v;
  }
  __syncthreads();
}

__global__ void __launch_bounds__(128) k_former(
    const bf16* __restrict__ h3, const float* __restrict__ scale3,
    const float* __restrict__ bias3, const bf16* __restrict__ psi,
    const bf16* __restrict__ wsT, const float* __restrict__ wb,
    const float* __restrict__ alpha, const bf16* __restrict__ wqkvT,
    const bf16* __restrict__ woT, const float* __restrict__ ln1g,
    const float* __restrict__ ln1b, const float* __restrict__ ln2g,
    const float* __restrict__ ln2b, float* __restrict__ out) {
  __shared__ float sH[K_ * HS];                      // hcur f32
  __shared__ __align__(16) bf16 sHb[K_ * HBS];       // hcur bf16 (A-operand)
  __shared__ __align__(16) bf16 sX[K_ * HBS];        // t1 / q / o staging
  __shared__ __align__(16) bf16 sK[K_ * HBS];        // keys
  __shared__ __align__(16) bf16 sVT[C_ * VTS];       // v^T: [h*32+d][token]
  __shared__ __align__(16) bf16 sScr[H_ * K_ * VTS]; // psi (5x32x32) | scores
  __shared__ float sMean[K_], sRstd[K_];

  int p = blockIdx.x;
  int tid = threadIdx.x, wave = tid >> 5, lane = tid & 31;

  // Load pf = relu(bn3(h3)) into resident state
  for (int idx = tid; idx < K_ * C_; idx += 128) {
    int r = idx >> 7, c = idx & 127;
    float v = (float)h3[(size_t)p * (K_ * C_) + idx];
    v = fmaxf(v * scale3[c] + bias3[c], 0.f);
    sH[r * HS + c] = v;
    sHb[r * HBS + c] = (bf16)v;
  }
  __syncthreads();

  for (int d = 0; d < DEPTH_; ++d) {
    // ---- async-stage all 5 psi matrices for this patch into sScr ----------
    {
      const bf16* psiG = psi + (size_t)p * (J_ * 1024);
      for (int ch = tid; ch < (J_ * 1024) / 8; ch += 128) {   // 640 chunks
        uint32_t lo = (uint32_t)(uintptr_t)(&sScr[ch * 8]);
        uint64_t ga = (uint64_t)(uintptr_t)(psiG + ch * 8);
        async_ld_b128(lo, ga);
      }
      wait_async0();
      __syncthreads();
    }
    // ---- wavelet mixing: mix = sum_j alpha[d,j] * (psi_j @ hcur) @ ws[d,j]
    v8f mixAcc[4];
#pragma unroll
    for (int i = 0; i < 4; ++i) mixAcc[i] = (v8f){};
    for (int j = 0; j < J_; ++j) {
      const bf16* psiL = sScr + j * 1024;
      // t1 = psi_j @ hcur  (A from LDS psi, B = hcur^T gathered from sHb)
      for (int t2 = wave; t2 < 16; t2 += 4) {
        int mt = t2 & 1, nt = t2 >> 1;
        v16bf a = ldA16(psiL, 32, mt * 16, 0, lane);
        v16bf bm = ldB16_trans(sHb, HBS, nt * 16, 0, lane);
        v8f c0 = {};
        c0 = wmma_bf16(a, bm, c0);
        int col = nt * 16 + (lane & 15);
        int g = lane >> 4;
#pragma unroll
        for (int r = 0; r < 8; ++r)
          sX[(mt * 16 + r + 8 * g) * HBS + col] = (bf16)c0[r];
      }
      __syncthreads();
      // mixAcc += alpha * (t1 @ ws[d][j])
      float aj = alpha[d * J_ + j];
      const bf16* wsP = wsT + ((size_t)(d * J_ + j)) * C_ * C_;
      if (j + 1 < J_)
        __builtin_prefetch(wsP + (size_t)C_ * C_, 0, 1);
#pragma unroll
      for (int i = 0; i < 4; ++i) {
        int t2 = wave + 4 * i;
        int mt = t2 & 1, nt = t2 >> 1;
        v8f c0 = {};
        for (int kt = 0; kt < 4; ++kt) {
          v16bf a = ldA16(sX, HBS, mt * 16, kt * 32, lane);
          v16bf bm = ldB16(wsP, C_, nt * 16, kt * 32, lane);
          c0 = wmma_bf16(a, bm, c0);
        }
#pragma unroll
        for (int r = 0; r < 8; ++r) mixAcc[i][r] += aj * c0[r];
      }
      __syncthreads();
    }
    // hpre = hcur + relu(mix + wb[d]) (in place on sH)
#pragma unroll
    for (int i = 0; i < 4; ++i) {
      int t2 = wave + 4 * i;
      int mt = t2 & 1, nt = t2 >> 1;
      int col = nt * 16 + (lane & 15);
      int g = lane >> 4;
      float wbv = wb[d * C_ + col];
#pragma unroll
      for (int r = 0; r < 8; ++r) {
        int m = mt * 16 + r + 8 * g;
        sH[m * HS + col] += fmaxf(mixAcc[i][r] + wbv, 0.f);
      }
    }
    __syncthreads();
    layer_norm(sH, sHb, sMean, sRstd, ln1g + d * C_, ln1b + d * C_, tid);

    // ---- qkv = hcur @ wqkv[d] : 48 output tiles, q pre-scaled by 1/sqrt(dh)
    const bf16* wqP = wqkvT + (size_t)d * 3 * C_ * C_;
    for (int t2 = wave; t2 < 48; t2 += 4) {
      int mt = t2 & 1, nt = t2 >> 1;
      v8f c0 = {};
      for (int kt = 0; kt < 4; ++kt) {
        v16bf a = ldA16(sHb, HBS, mt * 16, kt * 32, lane);
        v16bf bm = ldB16(wqP, C_, nt * 16, kt * 32, lane);
        c0 = wmma_bf16(a, bm, c0);
      }
      int col = nt * 16 + (lane & 15);
      int g = lane >> 4;
#pragma unroll
      for (int r = 0; r < 8; ++r) {
        int m = mt * 16 + r + 8 * g;
        float v = c0[r];
        if (col < C_)
          sX[m * HBS + col] = (bf16)(v * 0.17677669529663687f);
        else if (col < 2 * C_)
          sK[m * HBS + (col - C_)] = (bf16)v;
        else
          sVT[(col - 2 * C_) * VTS + m] = (bf16)v;
      }
    }
    __syncthreads();
    // ---- scores per head: s = q_h @ k_h^T  (into sScr, psi no longer needed)
    for (int t2 = wave; t2 < 16; t2 += 4) {
      int h = t2 >> 2, mt = (t2 >> 1) & 1, nt = t2 & 1;
      v16bf a = ldA16(sX, HBS, mt * 16, h * 32, lane);
      v16bf bm = ldB16(sK, HBS, nt * 16, h * 32, lane);
      v8f c0 = {};
      c0 = wmma_bf16(a, bm, c0);
      int col = nt * 16 + (lane & 15);
      int g = lane >> 4;
#pragma unroll
      for (int r = 0; r < 8; ++r)
        sScr[(h * 32 + mt * 16 + r + 8 * g) * VTS + col] = (bf16)c0[r];
    }
    __syncthreads();
    // ---- softmax over l, one (h,k) row per thread, in place in sScr
    {
      bf16* rowp = sScr + tid * VTS;
      float mx = -1e30f;
#pragma unroll
      for (int l2 = 0; l2 < K_; ++l2) mx = fmaxf(mx, (float)rowp[l2]);
      float sum = 0.f;
      float e[K_];
#pragma unroll
      for (int l2 = 0; l2 < K_; ++l2) {
        float ev = __expf((float)rowp[l2] - mx);
        e[l2] = ev;
        sum += ev;
      }
      float inv = 1.f / sum;
#pragma unroll
      for (int l2 = 0; l2 < K_; ++l2) rowp[l2] = (bf16)(e[l2] * inv);
    }
    __syncthreads();
    // ---- o = att @ v (per head), staged into sX
    for (int t2 = wave; t2 < 16; t2 += 4) {
      int h = t2 >> 2, mt = (t2 >> 1) & 1, nt = t2 & 1;
      v16bf a = ldA16(sScr + h * 32 * VTS, VTS, mt * 16, 0, lane);
      v16bf bm = ldB16(sVT + h * 32 * VTS, VTS, nt * 16, 0, lane);
      v8f c0 = {};
      c0 = wmma_bf16(a, bm, c0);
      int col = h * 32 + nt * 16 + (lane & 15);
      int g = lane >> 4;
#pragma unroll
      for (int r = 0; r < 8; ++r)
        sX[(mt * 16 + r + 8 * g) * HBS + col] = (bf16)c0[r];
    }
    __syncthreads();
    // ---- out projection + residual (in place on sH)
    const bf16* woP = woT + (size_t)d * C_ * C_;
    for (int t2 = wave; t2 < 16; t2 += 4) {
      int mt = t2 & 1, nt = t2 >> 1;
      v8f c0 = {};
      for (int kt = 0; kt < 4; ++kt) {
        v16bf a = ldA16(sX, HBS, mt * 16, kt * 32, lane);
        v16bf bm = ldB16(woP, C_, nt * 16, kt * 32, lane);
        c0 = wmma_bf16(a, bm, c0);
      }
      int col = nt * 16 + (lane & 15);
      int g = lane >> 4;
#pragma unroll
      for (int r = 0; r < 8; ++r) {
        int m = mt * 16 + r + 8 * g;
        sH[m * HS + col] += c0[r];
      }
    }
    __syncthreads();
    layer_norm(sH, sHb, sMean, sRstd, ln2g + d * C_, ln2b + d * C_, tid);
  }

  // ---- max-pool over K tokens
  if (tid < C_) {
    float mx = -1e30f;
#pragma unroll
    for (int r = 0; r < K_; ++r) mx = fmaxf(mx, sH[r * HS + tid]);
    out[(size_t)p * C_ + tid] = mx;
  }
}

// ---------------------------------------------------------------------------
// Workspace layout (bytes)
// ---------------------------------------------------------------------------
static constexpr size_t O_FPS  = 0;
static constexpr size_t O_KNN  = O_FPS + (size_t)B_ * S_ * 4;
static constexpr size_t O_GN   = O_KNN + (size_t)ROWS_ * 4;
static constexpr size_t O_H0   = O_GN + (size_t)ROWS_ * 3 * 4;
static constexpr size_t O_H1   = O_H0 + (size_t)ROWS_ * CINP_ * 2;
static constexpr size_t O_H2   = O_H1 + (size_t)ROWS_ * 64 * 2;
static constexpr size_t O_H3   = O_H2 + (size_t)ROWS_ * 64 * 2;
static constexpr size_t O_PSI  = O_H3 + (size_t)ROWS_ * 128 * 2;
static constexpr size_t O_PART = O_PSI + (size_t)P_ * J_ * 1024 * 2;  // partials
static constexpr size_t O_BN   = O_PART + (size_t)NBLK_ * 128 * 2 * 4;
static constexpr size_t O_W1B  = O_BN + 512 * 4;
static constexpr size_t O_W2B  = O_W1B + (size_t)64 * CINP_ * 2;
static constexpr size_t O_W3B  = O_W2B + (size_t)64 * 64 * 2;
static constexpr size_t O_WST  = O_W3B + (size_t)128 * 64 * 2;
static constexpr size_t O_WQT  = O_WST + (size_t)DEPTH_ * J_ * C_ * C_ * 2;
static constexpr size_t O_WOT  = O_WQT + (size_t)DEPTH_ * 3 * C_ * C_ * 2;

extern "C" void kernel_launch(void* const* d_in, const int* in_sizes, int n_in,
                              void* d_out, int out_size, void* d_ws,
                              size_t ws_size, hipStream_t stream) {
  (void)in_sizes; (void)n_in; (void)out_size; (void)ws_size;
  const float* xyz    = (const float*)d_in[0];
  const float* points = (const float*)d_in[1];
  const float* w1     = (const float*)d_in[2];
  const float* g1     = (const float*)d_in[3];
  const float* b1     = (const float*)d_in[4];
  const float* w2     = (const float*)d_in[5];
  const float* g2     = (const float*)d_in[6];
  const float* b2     = (const float*)d_in[7];
  const float* w3     = (const float*)d_in[8];
  const float* g3     = (const float*)d_in[9];
  const float* b3     = (const float*)d_in[10];
  const float* wsW    = (const float*)d_in[11];
  const float* wb     = (const float*)d_in[12];
  const float* alpha  = (const float*)d_in[13];
  const float* wqkv   = (const float*)d_in[14];
  const float* wo     = (const float*)d_in[15];
  const float* ln1g   = (const float*)d_in[16];
  const float* ln1b   = (const float*)d_in[17];
  const float* ln2g   = (const float*)d_in[18];
  const float* ln2b   = (const float*)d_in[19];

  char* wsb = (char*)d_ws;
  int*   fpsIdx = (int*)(wsb + O_FPS);
  int*   knn    = (int*)(wsb + O_KNN);
  float* gnorm  = (float*)(wsb + O_GN);
  bf16*  h0     = (bf16*)(wsb + O_H0);
  bf16*  h1     = (bf16*)(wsb + O_H1);
  bf16*  h2     = (bf16*)(wsb + O_H2);
  bf16*  h3     = (bf16*)(wsb + O_H3);
  bf16*  psi    = (bf16*)(wsb + O_PSI);
  float* psum   = (float*)(wsb + O_PART);
  float* psq    = psum + (size_t)NBLK_ * 128;
  float* bn     = (float*)(wsb + O_BN);
  float* scale1 = bn + 0,   *bias1 = bn + 64;
  float* scale2 = bn + 128, *bias2 = bn + 192;
  float* scale3 = bn + 256, *bias3 = bn + 384;
  bf16*  w1b    = (bf16*)(wsb + O_W1B);
  bf16*  w2b    = (bf16*)(wsb + O_W2B);
  bf16*  w3b    = (bf16*)(wsb + O_W3B);
  bf16*  wsT    = (bf16*)(wsb + O_WST);
  bf16*  wqkvT  = (bf16*)(wsb + O_WQT);
  bf16*  woT    = (bf16*)(wsb + O_WOT);

  float* newXyz = (float*)d_out;              // B*S*3 = 24576 floats
  float* outFeat = (float*)d_out + (size_t)B_ * S_ * 3;

  // Weight prep (bf16, padded / [out][in] transposed)
  k_prep_pad<<<(64 * CINP_ + 255) / 256, 256, 0, stream>>>(w1, w1b, 64, 67, CINP_);
  k_prep_pad<<<(64 * 64 + 255) / 256, 256, 0, stream>>>(w2, w2b, 64, 64, 64);
  k_prep_pad<<<(128 * 64 + 255) / 256, 256, 0, stream>>>(w3, w3b, 128, 64, 64);
  k_prep_T<<<(DEPTH_ * J_ * C_ * C_ + 255) / 256, 256, 0, stream>>>(
      wsW, wsT, DEPTH_ * J_, C_, C_);
  k_prep_T<<<(DEPTH_ * C_ * 3 * C_ + 255) / 256, 256, 0, stream>>>(
      wqkv, wqkvT, DEPTH_, C_, 3 * C_);
  k_prep_T<<<(DEPTH_ * C_ * C_ + 255) / 256, 256, 0, stream>>>(wo, woT, DEPTH_,
                                                               C_, C_);

  // Sampling / grouping
  k_fps<<<B_, 1024, 0, stream>>>(xyz, fpsIdx, newXyz);
  k_knn<<<P_, 256, 0, stream>>>(xyz, newXyz, knn);
  k_gather<<<(ROWS_ + 255) / 256, 256, 0, stream>>>(xyz, points, newXyz, knn,
                                                    h0, gnorm);

  // MLP with global BN stats between layers
  k_gemm<<<ROWS_ / 32, 128, 0, stream>>>(h0, CINP_, w1b, 64, 3, h1, 0, nullptr,
                                         nullptr);
  k_stats<<<NBLK_, 256, 0, stream>>>(h1, ROWS_, 64, psum, psq);
  k_bnparam<<<1, 128, 0, stream>>>(psum, psq, NBLK_, g1, b1, 64,
                                   1.f / (float)ROWS_, scale1, bias1);
  k_gemm<<<ROWS_ / 32, 128, 0, stream>>>(h1, 64, w2b, 64, 2, h2, 1, scale1,
                                         bias1);
  k_stats<<<NBLK_, 256, 0, stream>>>(h2, ROWS_, 64, psum, psq);
  k_bnparam<<<1, 128, 0, stream>>>(psum, psq, NBLK_, g2, b2, 64,
                                   1.f / (float)ROWS_, scale2, bias2);
  k_gemm<<<ROWS_ / 32, 128, 0, stream>>>(h2, 64, w3b, 128, 2, h3, 1, scale2,
                                         bias2);
  k_stats<<<NBLK_, 256, 0, stream>>>(h3, ROWS_, 128, psum, psq);
  k_bnparam<<<1, 128, 0, stream>>>(psum, psq, NBLK_, g3, b3, 128,
                                   1.f / (float)ROWS_, scale3, bias3);

  // Wavelet operators + transformer + pooling
  k_psi<<<P_, dim3(32, 32), 0, stream>>>(gnorm, psi);
  k_former<<<P_, 128, 0, stream>>>(h3, scale3, bias3, psi, wsT, wb, alpha,
                                   wqkvT, woT, ln1g, ln1b, ln2g, ln2b, outFeat);
}
